// SConv_78829829751211
// MI455X (gfx1250) — matrix operand order repaired and checked by reading, not compiled, not run on hardware
//
#include <hip/hip_runtime.h>
#include <hip/hip_bf16.h>

typedef __attribute__((ext_vector_type(16))) _Float16 v16h;
typedef __attribute__((ext_vector_type(8)))  _Float16 v8h;
typedef __attribute__((ext_vector_type(8)))  float    v8f;

static constexpr int Bb = 4, Ll = 4096, Dd = 512, Hh = 1024;
static constexpr int NC = 8, CL = Ll / NC;        // 8 chunks of 512
static constexpr int MM = Bb * Ll;                // 16384 rows

__device__ __forceinline__ v16h cat8(v8h lo, v8h hi) {
  return __builtin_shufflevector(lo, hi, 0,1,2,3,4,5,6,7,8,9,10,11,12,13,14,15);
}

__device__ __forceinline__ v8f wmma_f16(v16h a, v16h b, v8f c) {
  return __builtin_amdgcn_wmma_f32_16x16x32_f16(false, a, false, b, (short)0, c,
                                                false, false);
}

// ---------------- pack / prep kernels ----------------

__global__ __launch_bounds__(256) void sconv_pack_x(const float* __restrict__ in,
                                                    _Float16* __restrict__ out, int n) {
  int i = blockIdx.x * 256 + threadIdx.x;
  if (i < n) out[i] = (_Float16)in[i];
}

// out[c*R + r] = sgn * in[r*C + c]   (transpose [R,C] -> [C,R], f32 -> f16)
__global__ __launch_bounds__(256) void sconv_pack_wt(const float* __restrict__ in,
                                                     _Float16* __restrict__ out,
                                                     int R, int C, float sgn) {
  int i = blockIdx.x * 256 + threadIdx.x;
  if (i >= R * C) return;
  int r = i / C, c = i % C;
  out[(size_t)c * R + r] = (_Float16)(in[i] * sgn);
}

// p = phazor / max(|phazor|, 1)
__global__ __launch_bounds__(256) void sconv_pnorm(const float* __restrict__ phr,
                                                   const float* __restrict__ phi,
                                                   float* __restrict__ pr,
                                                   float* __restrict__ pi) {
  int h = blockIdx.x * 256 + threadIdx.x;
  if (h >= Hh) return;
  float r = phr[h], im = phi[h];
  float d = fmaxf(sqrtf(r * r + im * im), 1.0f);
  pr[h] = r / d;
  pi[h] = im / d;
}

// ---------------- GEMM 1: xh = x @ W_in^T (dual-B: real & imag) ----------------
// A: [MM, Dd] f16 row-major. Br/Bi: [Dd, Hh] f16 row-major (pre-transposed w_in).
// Cr/Ci: [MM, Hh] f16. WG = 8 waves, tile 64(M) x 128(N); wave tile 32x32.

__global__ __launch_bounds__(256) void sconv_gemm_in(
    const _Float16* __restrict__ A,
    const _Float16* __restrict__ Br, const _Float16* __restrict__ Bi,
    _Float16* __restrict__ Cr, _Float16* __restrict__ Ci) {
  const int K = Dd, N = Hh;
  int tid = threadIdx.x;
  int lane = tid & 31, wid = tid >> 5;
  int wm = wid & 1, wn = wid >> 1;
  int m0 = blockIdx.x * 64 + wm * 32;
  int n0 = blockIdx.y * 128 + wn * 32;
  int lhalf = lane >> 4, lmod = lane & 15;
  int kb = lhalf * 8;   // A frag K sub-offset per ISA 16-bit A layout

  v8f acc_r[2][2] = {}; v8f acc_i[2][2] = {};

  for (int kk = 0; kk < K; kk += 32) {
    v16h a[2], br[2], bi[2];
#pragma unroll
    for (int mf = 0; mf < 2; ++mf) {
      const _Float16* ap = A + (size_t)(m0 + mf * 16 + lmod) * K + kk + kb;
      a[mf] = cat8(*(const v8h*)ap, *(const v8h*)(ap + 16));
    }
    const size_t brow = (size_t)(kk + kb) * N + n0 + lmod;   // B: lane <-> N, elements <-> K
#pragma unroll
    for (int nf = 0; nf < 2; ++nf) {
#pragma unroll
      for (int s = 0; s < 16; ++s) {
        size_t o = brow + (size_t)(s < 8 ? s : s + 8) * N + nf * 16;
        br[nf][s] = Br[o];
        bi[nf][s] = Bi[o];
      }
    }
    if (kk + 32 < K) {
      __builtin_prefetch(A + (size_t)(m0 + lmod) * K + kk + 32 + kb, 0, 1);
      __builtin_prefetch(Br + (size_t)(kk + 32 + lane) * N + n0, 0, 1);
      __builtin_prefetch(Bi + (size_t)(kk + 32 + lane) * N + n0, 0, 1);
    }
#pragma unroll
    for (int mf = 0; mf < 2; ++mf)
#pragma unroll
      for (int nf = 0; nf < 2; ++nf) {
        acc_r[mf][nf] = wmma_f16(a[mf], br[nf], acc_r[mf][nf]);
        acc_i[mf][nf] = wmma_f16(a[mf], bi[nf], acc_i[mf][nf]);
      }
  }
#pragma unroll
  for (int mf = 0; mf < 2; ++mf)
#pragma unroll
    for (int nf = 0; nf < 2; ++nf)
#pragma unroll
      for (int r = 0; r < 8; ++r) {
        size_t row = (size_t)(m0 + mf * 16 + lhalf * 8 + r);
        size_t col = (size_t)(n0 + nf * 16 + lmod);
        Cr[row * N + col] = (_Float16)acc_r[mf][nf][r];
        Ci[row * N + col] = (_Float16)acc_i[mf][nf][r];
      }
}

// ---------------- scan: c'[t] = p*c'[t-1] + xh[t], c'[-1] = lc ----------------

// phase 1: per-chunk carry with zero init (locals recomputed in phase 3)
__global__ __launch_bounds__(256) void sconv_scan_carry(
    const _Float16* __restrict__ xr, const _Float16* __restrict__ xi,
    const float* __restrict__ pr, const float* __restrict__ pi,
    float* __restrict__ car, float* __restrict__ cai) {
  int idx = blockIdx.x * 256 + threadIdx.x;   // over B*NC*H
  int h = idx & (Hh - 1);
  int bc = idx >> 10;
  int c = bc & (NC - 1);
  int b = bc >> 3;
  float Pr = pr[h], Pi = pi[h];
  float sr = 0.f, si = 0.f;
  size_t base = ((size_t)b * Ll + (size_t)c * CL) * Hh + h;
  for (int t = 0; t < CL; ++t) {
    float xr_ = (float)xr[base], xi_ = (float)xi[base];
    float nr = fmaf(Pr, sr, fmaf(-Pi, si, xr_));
    float ni = fmaf(Pr, si, fmaf(Pi, sr, xi_));
    sr = nr; si = ni;
    base += Hh;
  }
  car[idx] = sr; cai[idx] = si;
}

// phase 2: scan the NC carries per (b,h) with multiplier p^CL, init state = lc
__global__ __launch_bounds__(256) void sconv_scan_combine(
    const float* __restrict__ car, const float* __restrict__ cai,
    const float* __restrict__ pr, const float* __restrict__ pi,
    const float* __restrict__ lcr, const float* __restrict__ lci,
    float* __restrict__ str, float* __restrict__ sti) {
  int idx = blockIdx.x * 256 + threadIdx.x;   // over B*H
  if (idx >= Bb * Hh) return;
  int h = idx & (Hh - 1);
  int b = idx >> 10;
  float qr = pr[h], qi = pi[h];
#pragma unroll
  for (int i = 0; i < 9; ++i) {               // p^512 via 9 squarings
    float nr = qr * qr - qi * qi, ni = 2.f * qr * qi;
    qr = nr; qi = ni;
  }
  float sr = lcr[h], si = lci[h];
#pragma unroll
  for (int c = 0; c < NC; ++c) {
    size_t o = ((size_t)b * NC + c) * Hh + h;
    str[o] = sr; sti[o] = si;                 // state entering chunk c
    float nr = fmaf(qr, sr, fmaf(-qi, si, car[o]));
    float ni = fmaf(qr, si, fmaf(qi, sr, cai[o]));
    sr = nr; si = ni;
  }
}

// phase 3: recompute local scan with correct entry state; overwrite xh -> conv
__global__ __launch_bounds__(256) void sconv_scan_apply(
    _Float16* __restrict__ xr, _Float16* __restrict__ xi,
    const float* __restrict__ pr, const float* __restrict__ pi,
    const float* __restrict__ str, const float* __restrict__ sti) {
  int idx = blockIdx.x * 256 + threadIdx.x;   // over B*NC*H
  int h = idx & (Hh - 1);
  int bc = idx >> 10;
  int c = bc & (NC - 1);
  int b = bc >> 3;
  float Pr = pr[h], Pi = pi[h];
  float sr = str[idx], si = sti[idx];
  size_t base = ((size_t)b * Ll + (size_t)c * CL) * Hh + h;
  for (int t = 0; t < CL; ++t) {
    float xr_ = (float)xr[base], xi_ = (float)xi[base];
    float nr = fmaf(Pr, sr, fmaf(-Pi, si, xr_));
    float ni = fmaf(Pr, si, fmaf(Pi, sr, xi_));
    sr = nr; si = ni;
    xr[base] = (_Float16)sr;
    xi[base] = (_Float16)si;
    base += Hh;
  }
}

// ---------------- GEMM 2: y = cr @ w_out_r^T - ci @ w_out_i^T ----------------
// Two K-passes (real/imag), f32 output. B matrices pre-transposed to [Hh, Dd],
// with the minus sign folded into B1.

__global__ __launch_bounds__(256) void sconv_gemm_out(
    const _Float16* __restrict__ A0, const _Float16* __restrict__ B0,
    const _Float16* __restrict__ A1, const _Float16* __restrict__ B1,
    float* __restrict__ C) {
  const int K = Hh, N = Dd;
  int tid = threadIdx.x;
  int lane = tid & 31, wid = tid >> 5;
  int wm = wid & 1, wn = wid >> 1;
  int m0 = blockIdx.x * 64 + wm * 32;
  int n0 = blockIdx.y * 128 + wn * 32;
  int lhalf = lane >> 4, lmod = lane & 15;
  int kb = lhalf * 8;

  v8f acc[2][2] = {};

#pragma unroll
  for (int pass = 0; pass < 2; ++pass) {
    const _Float16* A  = pass ? A1 : A0;
    const _Float16* Bm = pass ? B1 : B0;
    for (int kk = 0; kk < K; kk += 32) {
      v16h a[2], bfr[2];
#pragma unroll
      for (int mf = 0; mf < 2; ++mf) {
        const _Float16* ap = A + (size_t)(m0 + mf * 16 + lmod) * K + kk + kb;
        a[mf] = cat8(*(const v8h*)ap, *(const v8h*)(ap + 16));
      }
      const size_t brow = (size_t)(kk + kb) * N + n0 + lmod;
#pragma unroll
      for (int nf = 0; nf < 2; ++nf)
#pragma unroll
        for (int s = 0; s < 16; ++s)
          bfr[nf][s] = Bm[brow + (size_t)(s < 8 ? s : s + 8) * N + nf * 16];
      if (kk + 32 < K) {
        __builtin_prefetch(A + (size_t)(m0 + lmod) * K + kk + 32 + kb, 0, 1);
        __builtin_prefetch(Bm + (size_t)(kk + 32 + lane) * N + n0, 0, 1);
      }
#pragma unroll
      for (int mf = 0; mf < 2; ++mf)
#pragma unroll
        for (int nf = 0; nf < 2; ++nf)
          acc[mf][nf] = wmma_f16(a[mf], bfr[nf], acc[mf][nf]);
    }
  }
#pragma unroll
  for (int mf = 0; mf < 2; ++mf)
#pragma unroll
    for (int nf = 0; nf < 2; ++nf)
#pragma unroll
      for (int r = 0; r < 8; ++r) {
        size_t row = (size_t)(m0 + mf * 16 + lhalf * 8 + r);
        size_t col = (size_t)(n0 + nf * 16 + lmod);
        C[row * N + col] = acc[mf][nf][r];
      }
}

// ---------------- launch ----------------

extern "C" void kernel_launch(void* const* d_in, const int* in_sizes, int n_in,
                              void* d_out, int out_size, void* d_ws, size_t ws_size,
                              hipStream_t stream) {
  (void)in_sizes; (void)n_in; (void)out_size; (void)ws_size;
  const float* x       = (const float*)d_in[0];
  const float* w_in_r  = (const float*)d_in[1];
  const float* w_in_i  = (const float*)d_in[2];
  const float* w_out_r = (const float*)d_in[3];
  const float* w_out_i = (const float*)d_in[4];
  const float* phr     = (const float*)d_in[5];
  const float* phi     = (const float*)d_in[6];
  const float* lcr     = (const float*)d_in[7];
  const float* lci     = (const float*)d_in[8];
  float* y = (float*)d_out;

  char* w = (char*)d_ws;
  size_t cur = 0;
  auto take = [&](size_t bytes) -> void* {
    void* p = w + cur;
    cur += (bytes + 255) & ~(size_t)255;
    return p;
  };
  _Float16* x16     = (_Float16*)take((size_t)MM * Dd * 2);
  _Float16* wtin_r  = (_Float16*)take((size_t)Dd * Hh * 2);
  _Float16* wtin_i  = (_Float16*)take((size_t)Dd * Hh * 2);
  _Float16* wtout_r = (_Float16*)take((size_t)Hh * Dd * 2);
  _Float16* wtout_i = (_Float16*)take((size_t)Hh * Dd * 2);
  _Float16* xh_r    = (_Float16*)take((size_t)MM * Hh * 2);  // becomes conv_r
  _Float16* xh_i    = (_Float16*)take((size_t)MM * Hh * 2);  // becomes conv_i
  float* pn_r  = (float*)take(Hh * 4);
  float* pn_i  = (float*)take(Hh * 4);
  float* car_r = (float*)take((size_t)Bb * NC * Hh * 4);
  float* car_i = (float*)take((size_t)Bb * NC * Hh * 4);
  float* st_r  = (float*)take((size_t)Bb * NC * Hh * 4);
  float* st_i  = (float*)take((size_t)Bb * NC * Hh * 4);

  // prep
  sconv_pack_x<<<(MM * Dd) / 256, 256, 0, stream>>>(x, x16, MM * Dd);
  sconv_pack_wt<<<(Hh * Dd) / 256, 256, 0, stream>>>(w_in_r, wtin_r, Hh, Dd, 1.0f);
  sconv_pack_wt<<<(Hh * Dd) / 256, 256, 0, stream>>>(w_in_i, wtin_i, Hh, Dd, 1.0f);
  sconv_pack_wt<<<(Hh * Dd) / 256, 256, 0, stream>>>(w_out_r, wtout_r, Dd, Hh, 1.0f);
  sconv_pack_wt<<<(Hh * Dd) / 256, 256, 0, stream>>>(w_out_i, wtout_i, Dd, Hh, -1.0f);
  sconv_pnorm<<<Hh / 256, 256, 0, stream>>>(phr, phi, pn_r, pn_i);

  // xh = x @ W_in^T  (real & imag)
  sconv_gemm_in<<<dim3(MM / 64, Hh / 128), 256, 0, stream>>>(x16, wtin_r, wtin_i,
                                                            xh_r, xh_i);
  // chunked complex scan (in-place xh -> conv_with_past)
  sconv_scan_carry<<<(Bb * NC * Hh) / 256, 256, 0, stream>>>(xh_r, xh_i, pn_r, pn_i,
                                                             car_r, car_i);
  sconv_scan_combine<<<(Bb * Hh) / 256, 256, 0, stream>>>(car_r, car_i, pn_r, pn_i,
                                                          lcr, lci, st_r, st_i);
  sconv_scan_apply<<<(Bb * NC * Hh) / 256, 256, 0, stream>>>(xh_r, xh_i, pn_r, pn_i,
                                                             st_r, st_i);
  // y = Re(conv @ W_out^T)
  sconv_gemm_out<<<dim3(MM / 64, Dd / 128), 256, 0, stream>>>(xh_r, wtout_r,
                                                              xh_i, wtout_i, y);
}
